// TimeAwareKAN_88776974009049
// MI455X (gfx1250) — compile-verified, matrix operand order checked by reading, not run on hardware
//
#include <hip/hip_runtime.h>
#include <hip/hip_bf16.h>

// ---------------------------------------------------------------------------
// TimeAwareKAN fused as one bf16 WMMA GEMM:
//   out[M=49152, O=512] = A[M,768] @ W[512,768]^T + base_b
//   A = [ silu(x_in) (K=0..127) | exp(-((x_in[i]-grid[g])*2)^2) (K=128..767, k=128+i*5+g) ]
//   W = [ base_w (O x 128)      | spline_weight (O x 128 x 5 contiguous) ]
// Threads 0..127 build A slab rows, threads 128..255 build B slab rows (2 each).
// Basis k-offset is grid-uniform so (k/5, k%5) fold to compile-time constants,
// and the 5 RBF values per x-column come from 2 exps via E*F^g*e^{-g^2}.
// ---------------------------------------------------------------------------

typedef __attribute__((ext_vector_type(16))) __bf16 v16bf;
typedef __attribute__((ext_vector_type(8)))  float  v8f;

union Frag {
    v16bf v;
    uint4 q[2];
};

union H32 {
    __bf16 h[32];
    uint4  q[4];
};

#define BDIM   256
#define BM     128
#define BN     256
#define BK     32
#define KTOT   768
#define NSTEPS (KTOT / BK)     // 24; steps 0..3 = silu region, 4..23 = basis region
#define CIN    120
#define CEMB   8
#define CXIN   128             // CIN + CEMB
#define OOUT   512
#define HH     24
#define NN     512
#define XS     132             // padded row stride (floats) for x tile in LDS
#define SLS    40              // padded row stride (bf16) for A/B slabs (32 + 8 pad)

// Store a 32-wide bf16 row (64 bytes) as 4x ds_store_b128.
__device__ __forceinline__ void store_row(__bf16* dst, const H32& r) {
    #pragma unroll
    for (int q = 0; q < 4; ++q)
        *(uint4*)(dst + q * 8) = r.q[q];
}

// Convert 32 contiguous f32 weights to a bf16 slab row.
__device__ __forceinline__ void weights_row(const float* __restrict__ wsrc,
                                            __bf16* __restrict__ dst) {
    H32 r;
    #pragma unroll
    for (int c4 = 0; c4 < 8; ++c4) {
        float4 w = *(const float4*)(wsrc + c4 * 4);
        r.h[c4 * 4 + 0] = (__bf16)w.x;
        r.h[c4 * 4 + 1] = (__bf16)w.y;
        r.h[c4 * 4 + 2] = (__bf16)w.z;
        r.h[c4 * 4 + 3] = (__bf16)w.w;
    }
    store_row(dst, r);
}

// RBF basis slab row covering kk2 = kk2b .. kk2b+31, with G0 = kk2b%5 compile
// time. For each x column i: exp(-(d0-g)^2) = E * F^g * e^{-g^2},
// E = exp(-d0^2), F = exp(2*d0), d0 = 2*v + 2  (grid_g = -1 + 0.5g, inv_h = 2).
template <int G0>
__device__ __forceinline__ void basis_row(const float* __restrict__ xr, int ii0,
                                          __bf16* __restrict__ dst) {
    H32 r;
    #pragma unroll
    for (int ioff = 0; ioff <= (G0 + 31) / 5; ++ioff) {   // 7-8 distinct columns
        float v  = xr[ii0 + ioff];
        float d0 = 2.0f * v + 2.0f;
        float E  = __expf(-d0 * d0);
        float F  = __expf(d0 + d0);
        float vals[5];
        float tp = E;
        vals[0] = tp;
        tp *= F; vals[1] = tp * 0.36787944117144233f;     // e^-1
        tp *= F; vals[2] = tp * 0.018315638888734179f;    // e^-4
        tp *= F; vals[3] = tp * 1.2340980408667956e-4f;   // e^-9
        tp *= F; vals[4] = tp * 1.1253517471925912e-7f;   // e^-16
        #pragma unroll
        for (int g = 0; g < 5; ++g) {
            const int c = ioff * 5 + g - G0;              // constant after unroll
            if (c >= 0 && c < 32) r.h[c] = (__bf16)vals[g];
        }
    }
    store_row(dst, r);
}

__global__ __launch_bounds__(BDIM, 2)
void takan_wmma_kernel(const float* __restrict__ x,
                       const long long* __restrict__ t_indices,
                       const float* __restrict__ time_emb,
                       const float* __restrict__ base_w,
                       const float* __restrict__ base_b,
                       const float* __restrict__ spline_w,
                       float* __restrict__ out) {
    __shared__ __align__(16) float  xlds[BM * XS];       // x_in tile, f32 [128][132]
    __shared__ __align__(16) __bf16 Aslab[BM * SLS];     // bf16 [128][40]
    __shared__ __align__(16) __bf16 Bslab[BN * SLS];     // bf16 [256][40]

    const int t    = threadIdx.x;
    const int lane = t & 31;
    const int wave = t >> 5;
    const int wr   = wave >> 2;        // 0..1  -> M half (4 subtiles)
    const int wc   = wave & 3;         // 0..3  -> N quarter (4 subtiles)
    const int lr   = lane & 15;        // fragment row within 16x16 subtile
    const int lk   = lane >> 4;        // fragment K-half selector

    const int nblk = blockIdx.x;       // 0..1   (O tiles of 256)
    const int mblk = blockIdx.y;       // 0..383 (M tiles)
    const int bh   = mblk >> 2;        // which (b,h) slab (N=512 = 4 tiles of 128)
    const int nt   = mblk & 3;
    const int hidx = bh % HH;

    // ---- stage contiguous 128x120 x tile into LDS (row stride XS) ----
    {
        const float4* xsrc =
            (const float4*)(x + (size_t)bh * (NN * CIN) + (size_t)nt * (BM * CIN));
        #pragma unroll
        for (int it = 0; it < (BM * CIN / 4) / BDIM; ++it) {   // 15 iters
            int idx  = t + it * BDIM;        // float4 index; 30 float4 per row
            int row  = idx / 30;
            int col4 = idx - row * 30;
            *(float4*)&xlds[row * XS + col4 * 4] = xsrc[idx];
        }
        // time embedding columns 120..127 (same h for whole tile)
        const int tix = (int)t_indices[hidx];
        {
            int row = t >> 1;
            int cb  = (t & 1) * 4;
            #pragma unroll
            for (int c = 0; c < 4; ++c)
                xlds[row * XS + CIN + cb + c] = time_emb[tix * CEMB + cb + c];
        }
    }
    __syncthreads();

    // ---- accumulators: 4 (M subtiles) x 4 (N subtiles) per wave ----
    v8f acc[4][4];
    const v8f vzero = {0.f, 0.f, 0.f, 0.f, 0.f, 0.f, 0.f, 0.f};
    #pragma unroll
    for (int i = 0; i < 4; ++i)
        #pragma unroll
        for (int j = 0; j < 4; ++j)
            acc[i][j] = vzero;

    // Role split (wave-uniform): waves 0-3 build A rows, waves 4-7 build B rows.
    const bool isA  = (t < BM);
    const int  rrow = isA ? t : (t - BM);

    #pragma unroll 1
    for (int kt = 0; kt < NSTEPS; ++kt) {
        // ---------- build A slab (activation) or B slab (weights, 2 rows) ----
        if (kt < 4) {
            const int kb = kt * BK;                       // uniform
            if (isA) {
                const float* xr = &xlds[rrow * XS];
                H32 r;
                #pragma unroll
                for (int c = 0; c < 32; ++c) {
                    float v = xr[kb + c];
                    r.h[c]  = (__bf16)__fdividef(v, 1.0f + __expf(-v));
                }
                store_row(&Aslab[rrow * SLS], r);
            } else {
                #pragma unroll
                for (int rr = 0; rr < 2; ++rr) {
                    const int nrow = rrow + rr * 128;
                    weights_row(base_w + (size_t)(nblk * BN + nrow) * CXIN + kb,
                                &Bslab[nrow * SLS]);
                }
            }
        } else {
            const int kk2b = (kt - 4) * BK;               // uniform, 0..608
            const int ii0  = kk2b / 5;                    // uniform
            if (isA) {
                const float* xr = &xlds[rrow * XS];
                __bf16* dst = &Aslab[rrow * SLS];
                switch (kk2b - ii0 * 5) {                 // uniform branch
                    case 0:  basis_row<0>(xr, ii0, dst); break;
                    case 1:  basis_row<1>(xr, ii0, dst); break;
                    case 2:  basis_row<2>(xr, ii0, dst); break;
                    case 3:  basis_row<3>(xr, ii0, dst); break;
                    default: basis_row<4>(xr, ii0, dst); break;
                }
            } else {
                #pragma unroll
                for (int rr = 0; rr < 2; ++rr) {
                    const int nrow = rrow + rr * 128;
                    weights_row(spline_w + (size_t)(nblk * BN + nrow) * (CXIN * 5) + kk2b,
                                &Bslab[nrow * SLS]);
                }
            }
        }
        __syncthreads();

        // ---------- load fragments (ISA 16-bit A/B layout) and WMMA ----------
        Frag fa[4], fb[4];
        #pragma unroll
        for (int i = 0; i < 4; ++i) {
            const __bf16* ap = &Aslab[((wr * 4 + i) * 16 + lr) * SLS + lk * 8];
            fa[i].q[0] = *(const uint4*)ap;          // K = lk*8 .. lk*8+7
            fa[i].q[1] = *(const uint4*)(ap + 16);   // K = 16+lk*8 .. 16+lk*8+7
        }
        #pragma unroll
        for (int j = 0; j < 4; ++j) {
            const __bf16* bp = &Bslab[((wc * 4 + j) * 16 + lr) * SLS + lk * 8];
            fb[j].q[0] = *(const uint4*)bp;
            fb[j].q[1] = *(const uint4*)(bp + 16);
        }
        #pragma unroll
        for (int i = 0; i < 4; ++i)
            #pragma unroll
            for (int j = 0; j < 4; ++j)
                acc[i][j] = __builtin_amdgcn_wmma_f32_16x16x32_bf16(
                    false, fa[i].v, false, fb[j].v,
                    (short)0, acc[i][j], false, false);
        __syncthreads();
    }

    // ---- epilogue: add bias, store f32 ----
    #pragma unroll
    for (int j = 0; j < 4; ++j) {
        const int col = nblk * BN + (wc * 4 + j) * 16 + lr;
        const float bb = base_b[col];
        #pragma unroll
        for (int i = 0; i < 4; ++i) {
            const int rowb = mblk * BM + (wr * 4 + i) * 16 + lk * 8;
            #pragma unroll
            for (int r = 0; r < 8; ++r)
                out[(size_t)(rowb + r) * OOUT + col] = acc[i][j][r] + bb;
        }
    }
}

extern "C" void kernel_launch(void* const* d_in, const int* in_sizes, int n_in,
                              void* d_out, int out_size, void* d_ws, size_t ws_size,
                              hipStream_t stream) {
    (void)in_sizes; (void)n_in; (void)d_ws; (void)ws_size; (void)out_size;
    const float*     x        = (const float*)d_in[0];
    const long long* t_ind    = (const long long*)d_in[1];
    const float*     time_emb = (const float*)d_in[2];
    const float*     base_w   = (const float*)d_in[3];
    const float*     base_b   = (const float*)d_in[4];
    const float*     spline_w = (const float*)d_in[5];
    float*           out      = (float*)d_out;

    dim3 grid(OOUT / BN, (4 * HH * NN) / BM);   // (2, 384)
    dim3 block(BDIM);
    takan_wmma_kernel<<<grid, block, 0, stream>>>(
        x, t_ind, time_emb, base_w, base_b, spline_w, out);
}
